// MDTA_59287728554589
// MI455X (gfx1250) — compile-verified, matrix operand order checked by reading, not run on hardware
//
#include <hip/hip_runtime.h>
#include <hip/hip_bf16.h>

// ---------------------------------------------------------------------------
// MDTA (Restormer channel attention) for MI455X / gfx1250.
// All GEMM-shaped stages use v_wmma_f32_16x16x32_bf16 (wave32 WMMA).
// LDS tiles are stored K-contiguous per fragment row (B tiles transposed) with
// 16B-aligned row strides: fragment loads are 2x ds_load_b128, staging stores
// are packed bf16x4 -> ds_store_b64 (4k x 4n micro-tiles for transposes).
// ---------------------------------------------------------------------------

typedef __attribute__((ext_vector_type(16))) __bf16 bf16x16;
typedef __attribute__((ext_vector_type(8)))  __bf16 bf16x8;
typedef __attribute__((ext_vector_type(4)))  __bf16 bf16x4;
typedef __attribute__((ext_vector_type(8)))  float  floatx8;

#define DIM       192
#define NHEADS    4
#define HDIM      48
#define HEIGHT    192
#define WIDTH     192
#define HW        (HEIGHT * WIDTH)      // 36864
#define BATCH     2
#define KC        32                    // WMMA K per step (bf16)
#define ATTN_KCH  96                    // K-chunks for q.k^T  (hw/(96*32)=12 steps)

__device__ __forceinline__ floatx8 wmma_bf16(bf16x16 a, bf16x16 b, floatx8 c) {
    // 8 args: (neg_a, A, neg_b, B, c_mod, C, reuse_a, reuse_b)
    return __builtin_amdgcn_wmma_f32_16x16x32_bf16(false, a, false, b,
                                                   (short)0, c, false, false);
}

// Fragment loader for the ISA 7.12.2 16-bit operand layout. `base` points at
// a [16 x strideElems] K-contiguous LDS tile; strideElems must be a multiple
// of 8 (16B) and base 16B-aligned => two aligned ds_load_b128 per lane.
__device__ __forceinline__ bf16x16 load_frag(const __bf16* base, int strideElems,
                                             int lane) {
    const int m  = lane & 15;
    const int kb = (lane >> 4) << 3;    // 0 or 8
    const __bf16* row = base + m * strideElems + kb;
    bf16x8 lo = *(const bf16x8*)(row);        // K = kb .. kb+7
    bf16x8 hi = *(const bf16x8*)(row + 16);   // K = kb+16 .. kb+23
    return __builtin_shufflevector(lo, hi, 0, 1, 2, 3, 4, 5, 6, 7,
                                   8, 9, 10, 11, 12, 13, 14, 15);
}

__device__ __forceinline__ bf16x4 pack4(float a, float b, float c, float d) {
    bf16x4 v = {(__bf16)a, (__bf16)b, (__bf16)c, (__bf16)d};
    return v;
}

// ---------------------------------------------------------------------------
// 1 & 8) Pointwise conv as GEMM:  Y[b][o][n] = sum_c W[o][c] * X[b][c][n]
// grid: (N/128, O/64, B), block 256 (8 waves: 4M x 2N).
// A staged [64][48] (M x K), B staged transposed [128][48] (N x K).
// ---------------------------------------------------------------------------
__global__ __launch_bounds__(256) void gemm_wmma_kernel(
    const float* __restrict__ W, const float* __restrict__ X,
    float* __restrict__ Y, int O, int K, int N)
{
    __shared__ __align__(16) __bf16 As[64][KC + 16];     // 64 x 48  (M x K)
    __shared__ __align__(16) __bf16 BsT[128][KC + 16];   // 128 x 48 (N x K)

    const int n0 = blockIdx.x * 128;
    const int o0 = blockIdx.y * 64;
    const int b  = blockIdx.z;
    X += (size_t)b * K * N;
    Y += (size_t)b * O * N;

    const int tid  = threadIdx.x;
    const int lane = tid & 31;
    const int wave = tid >> 5;
    const int wm   = wave >> 1;              // 0..3 : 16-row M slice
    const int wn   = wave & 1;               // 0..1 : 64-col N slice

    floatx8 acc[4] = {};

    for (int k0 = 0; k0 < K; k0 += KC) {
        // A tile: 64x32 floats = 512 quads, 2 per thread, bf16x4 LDS stores.
#pragma unroll
        for (int i = 0; i < 2; ++i) {
            int q = tid + i * 256;
            int r = q >> 3;                  // M row
            int c = (q & 7) * 4;             // K col
            const float4 f = *(const float4*)&W[(o0 + r) * K + (k0 + c)];
            *(bf16x4*)&As[r][c] = pack4(f.x, f.y, f.z, f.w);
        }
        // B tile: 32(k) x 128(n); each thread owns a 4k x 4n micro-tile:
        // 4 coalesced global_load_b128, 4 packed ds_store_b64 into BsT[n][k].
        {
            const int kr = (tid >> 5) * 4;   // 0..28
            const int nc = (tid & 31) * 4;   // 0..124
            const float* src = &X[(size_t)(k0 + kr) * N + (n0 + nc)];
            const float4 f0 = *(const float4*)(src);
            const float4 f1 = *(const float4*)(src + N);
            const float4 f2 = *(const float4*)(src + 2 * (size_t)N);
            const float4 f3 = *(const float4*)(src + 3 * (size_t)N);
            *(bf16x4*)&BsT[nc + 0][kr] = pack4(f0.x, f1.x, f2.x, f3.x);
            *(bf16x4*)&BsT[nc + 1][kr] = pack4(f0.y, f1.y, f2.y, f3.y);
            *(bf16x4*)&BsT[nc + 2][kr] = pack4(f0.z, f1.z, f2.z, f3.z);
            *(bf16x4*)&BsT[nc + 3][kr] = pack4(f0.w, f1.w, f2.w, f3.w);
        }
        // Prefetch next K-slab of the streaming operand (global_prefetch_b8).
        if (k0 + KC < K)
            __builtin_prefetch(&X[(size_t)(k0 + KC + (tid >> 5)) * N + n0 + (tid & 31) * 4], 0, 0);
        __syncthreads();

        bf16x16 a = load_frag(&As[wm * 16][0], KC + 16, lane);
#pragma unroll
        for (int t = 0; t < 4; ++t) {
            bf16x16 bf = load_frag(&BsT[wn * 64 + t * 16][0], KC + 16, lane);
            acc[t] = wmma_bf16(a, bf, acc[t]);
        }
        __syncthreads();
    }

    const int n_l  = lane & 15;
    const int m_hi = (lane >> 4) << 3;
#pragma unroll
    for (int t = 0; t < 4; ++t) {
        int ncol = n0 + wn * 64 + t * 16 + n_l;
#pragma unroll
        for (int r = 0; r < 8; ++r) {
            int m = o0 + wm * 16 + m_hi + r;
            Y[(size_t)m * N + ncol] = acc[t][r];
        }
    }
}

// ---------------------------------------------------------------------------
// 2) Depthwise 3x3, padding=1. grid: (HW/256, 3*DIM, B)
// ---------------------------------------------------------------------------
__global__ __launch_bounds__(256) void dwconv3x3_kernel(
    const float* __restrict__ in, const float* __restrict__ wdw,
    float* __restrict__ out, int C)
{
    int idx = blockIdx.x * 256 + threadIdx.x;
    int c = blockIdx.y, b = blockIdx.z;
    int y = idx / WIDTH, x = idx - y * WIDTH;
    const float* ip = in + ((size_t)b * C + c) * HW;
    const float* wp = wdw + c * 9;
    float s = 0.f;
#pragma unroll
    for (int dy = -1; dy <= 1; ++dy) {
        int yy = y + dy;
        if (yy < 0 || yy >= HEIGHT) continue;
#pragma unroll
        for (int dx = -1; dx <= 1; ++dx) {
            int xx = x + dx;
            if (xx < 0 || xx >= WIDTH) continue;
            s += ip[yy * WIDTH + xx] * wp[(dy + 1) * 3 + (dx + 1)];
        }
    }
    out[((size_t)b * C + c) * HW + idx] = s;
}

// ---------------------------------------------------------------------------
// 3) 1/L2-norm over hw for the 2*192 q/k channels. grid: (384, B), block 256.
// ---------------------------------------------------------------------------
__global__ __launch_bounds__(256) void l2norm_kernel(
    const float* __restrict__ qk, float* __restrict__ inv_norms)
{
    __shared__ float red[256];
    int c = blockIdx.x, b = blockIdx.y, t = threadIdx.x;
    const float* p = qk + ((size_t)b * (3 * DIM) + c) * HW;
    float s = 0.f;
    for (int i = t * 4; i < HW; i += 256 * 4) {
        const float4 f = *(const float4*)&p[i];
        s += f.x * f.x + f.y * f.y + f.z * f.z + f.w * f.w;
    }
    red[t] = s; __syncthreads();
    for (int st = 128; st > 0; st >>= 1) {
        if (t < st) red[t] += red[t + st];
        __syncthreads();
    }
    if (t == 0) {
        float n = sqrtf(red[0]);
        inv_norms[b * 384 + c] = 1.0f / fmaxf(n, 1e-12f);
    }
}

// ---------------------------------------------------------------------------
// 4) attn partials: per (kchunk, head, b), 9 waves cover the 3x3 16x16 tiles
// of the 48x48 output. attn = Qn * Kn^T; inv-norm fused into the bf16 stage.
// Both operands are naturally K-contiguous along hw.
// Deterministic: partials to [kchunk][b*4+h][48*48], reduced in fixed order.
// ---------------------------------------------------------------------------
__global__ __launch_bounds__(288) void attn_qk_kernel(
    const float* __restrict__ dwout, const float* __restrict__ inv_norms,
    float* __restrict__ partial)
{
    __shared__ __align__(16) __bf16 Qs[48][KC + 16];   // 48 x 48
    __shared__ __align__(16) __bf16 Ks[48][KC + 16];

    const int b = blockIdx.z, h = blockIdx.y;
    const float* qbase = dwout + ((size_t)b * (3 * DIM) + h * HDIM) * HW;
    const float* kbase = dwout + ((size_t)b * (3 * DIM) + DIM + h * HDIM) * HW;
    const float* qinv  = inv_norms + b * 384 + h * HDIM;
    const float* kinv  = inv_norms + b * 384 + DIM + h * HDIM;

    const int tid = threadIdx.x, lane = tid & 31, wave = tid >> 5;  // wave 0..8
    const int ti = wave / 3, tj = wave - 3 * ti;

    floatx8 acc = {};
    const int kelems = HW / ATTN_KCH;                 // 384
    const int kstart = blockIdx.x * kelems;

    for (int k0 = kstart; k0 < kstart + kelems; k0 += KC) {
        // 48 rows x 32 k per matrix = 384 quads; float4 load, bf16x4 store.
        for (int q = tid; q < 48 * (KC / 4); q += 288) {
            int r = q >> 3;
            int c = (q & 7) * 4;
            const float4 fq = *(const float4*)&qbase[(size_t)r * HW + k0 + c];
            const float4 fk = *(const float4*)&kbase[(size_t)r * HW + k0 + c];
            const float iq = qinv[r], ik = kinv[r];
            *(bf16x4*)&Qs[r][c] = pack4(fq.x * iq, fq.y * iq, fq.z * iq, fq.w * iq);
            *(bf16x4*)&Ks[r][c] = pack4(fk.x * ik, fk.y * ik, fk.z * ik, fk.w * ik);
        }
        __syncthreads();
        bf16x16 a  = load_frag(&Qs[ti * 16][0], KC + 16, lane);
        // B[k][n] = K[n][k]  ==> same K-contiguous gather from Ks rows.
        bf16x16 bf = load_frag(&Ks[tj * 16][0], KC + 16, lane);
        acc = wmma_bf16(a, bf, acc);
        __syncthreads();
    }

    float* pp = partial + ((size_t)blockIdx.x * (BATCH * NHEADS) + (b * NHEADS + h)) * (48 * 48);
    const int n_l = lane & 15, m_hi = (lane >> 4) << 3;
#pragma unroll
    for (int r = 0; r < 8; ++r)
        pp[(ti * 16 + m_hi + r) * 48 + tj * 16 + n_l] = acc[r];
}

// 5) fixed-order reduction of the K-chunk partials (float4 wide)
__global__ __launch_bounds__(256) void attn_reduce_kernel(
    const float* __restrict__ partial, float* __restrict__ attn, int n)
{
    int i = (blockIdx.x * 256 + threadIdx.x) * 4;
    if (i >= n) return;
    float4 s = make_float4(0.f, 0.f, 0.f, 0.f);
    for (int c = 0; c < ATTN_KCH; ++c) {
        const float4 f = *(const float4*)&partial[(size_t)c * n + i];
        s.x += f.x; s.y += f.y; s.z += f.z; s.w += f.w;
    }
    *(float4*)&attn[i] = s;
}

// ---------------------------------------------------------------------------
// 6) softmax over last dim with temperature = softplus(log_t) + 1e-6
// grid: (48, heads, B), block 64.
// ---------------------------------------------------------------------------
__global__ __launch_bounds__(64) void softmax_kernel(
    float* __restrict__ attn, const float* __restrict__ log_temp)
{
    __shared__ float buf[64];
    int d = blockIdx.x, h = blockIdx.y, b = blockIdx.z, t = threadIdx.x;
    float* p = attn + (((size_t)(b * NHEADS + h) * 48) + d) * 48;
    float temp = log1pf(expf(log_temp[h])) + 1e-6f;

    float v = (t < 48) ? p[t] * temp : -3.0e38f;
    buf[t] = v; __syncthreads();
    for (int s = 32; s > 0; s >>= 1) { if (t < s) buf[t] = fmaxf(buf[t], buf[t + s]); __syncthreads(); }
    float mx = buf[0]; __syncthreads();
    float e = (t < 48) ? expf(v - mx) : 0.f;
    buf[t] = e; __syncthreads();
    for (int s = 32; s > 0; s >>= 1) { if (t < s) buf[t] += buf[t + s]; __syncthreads(); }
    if (t < 48) p[t] = e / buf[0];
}

// ---------------------------------------------------------------------------
// 7) out = attn @ v : M=48 (zero-pad K 48->64), N tiled 128 wide.
// grid: (HW/128, heads, B), block 256 (8 waves, each: 3 M-tiles x 16 N-cols).
// A staged [48][80] (M x Kpad), V staged transposed [128][80] (N x Kpad).
// ---------------------------------------------------------------------------
__global__ __launch_bounds__(256) void attnv_kernel(
    const float* __restrict__ attn, const float* __restrict__ dwout,
    float* __restrict__ out)
{
    __shared__ __align__(16) __bf16 As[48][64 + 16];    // attn, K padded to 64
    __shared__ __align__(16) __bf16 VsT[128][64 + 16];  // v^T,  cols 48..63 zero

    const int b = blockIdx.z, h = blockIdx.y;
    const int n0 = blockIdx.x * 128;
    const float* ap = attn  + (size_t)(b * NHEADS + h) * 48 * 48;
    const float* vp = dwout + ((size_t)b * (3 * DIM) + 2 * DIM + h * HDIM) * HW;
    float* op       = out   + ((size_t)b * DIM + h * HDIM) * HW;

    const int tid = threadIdx.x, lane = tid & 31, wave = tid >> 5;

    // attn tile 48x64 (K-pad zeros): 768 quads, 3 per thread, bf16x4 stores.
#pragma unroll
    for (int i = 0; i < 3; ++i) {
        int q = tid + i * 256;
        int r = q >> 4, c = (q & 15) * 4;       // c = 0..60
        float4 f = make_float4(0.f, 0.f, 0.f, 0.f);
        if (c < 48) f = *(const float4*)&ap[r * 48 + c];
        *(bf16x4*)&As[r][c] = pack4(f.x, f.y, f.z, f.w);
    }
    // v tile 64(k) x 128(n): each thread 2x (4k x 4n) micro-tiles -> VsT[n][k].
#pragma unroll
    for (int i = 0; i < 2; ++i) {
        int u = tid + i * 256;
        int kr = (u >> 5) * 4;                  // 0..60
        int nc = (u & 31) * 4;
        float4 f0 = make_float4(0.f, 0.f, 0.f, 0.f), f1 = f0, f2 = f0, f3 = f0;
        if (kr < 48) {                          // kr multiple of 4: whole group valid
            const float* src = &vp[(size_t)kr * HW + n0 + nc];
            f0 = *(const float4*)(src);
            f1 = *(const float4*)(src + HW);
            f2 = *(const float4*)(src + 2 * (size_t)HW);
            f3 = *(const float4*)(src + 3 * (size_t)HW);
        }
        *(bf16x4*)&VsT[nc + 0][kr] = pack4(f0.x, f1.x, f2.x, f3.x);
        *(bf16x4*)&VsT[nc + 1][kr] = pack4(f0.y, f1.y, f2.y, f3.y);
        *(bf16x4*)&VsT[nc + 2][kr] = pack4(f0.z, f1.z, f2.z, f3.z);
        *(bf16x4*)&VsT[nc + 3][kr] = pack4(f0.w, f1.w, f2.w, f3.w);
    }
    __syncthreads();

    floatx8 acc[3] = {};
#pragma unroll
    for (int k0 = 0; k0 < 64; k0 += KC) {
#pragma unroll
        for (int t = 0; t < 3; ++t) {
            bf16x16 a  = load_frag(&As[t * 16][k0], 64 + 16, lane);
            bf16x16 bf = load_frag(&VsT[wave * 16][k0], 64 + 16, lane);
            acc[t] = wmma_bf16(a, bf, acc[t]);
        }
    }

    const int n_l = lane & 15, m_hi = (lane >> 4) << 3;
#pragma unroll
    for (int t = 0; t < 3; ++t)
#pragma unroll
        for (int r = 0; r < 8; ++r)
            op[(size_t)(t * 16 + m_hi + r) * HW + n0 + wave * 16 + n_l] = acc[t][r];
}

// ---------------------------------------------------------------------------
extern "C" void kernel_launch(void* const* d_in, const int* in_sizes, int n_in,
                              void* d_out, int out_size, void* d_ws, size_t ws_size,
                              hipStream_t stream) {
    const float* x        = (const float*)d_in[0];   // [2,192,192,192]
    const float* w_qkv    = (const float*)d_in[1];   // [576,192]
    const float* w_dw     = (const float*)d_in[2];   // [576,1,3,3]
    const float* w_proj   = (const float*)d_in[3];   // [192,192]
    const float* log_temp = (const float*)d_in[4];   // [4,1,1]
    float* y = (float*)d_out;

    // workspace layout (floats)
    float* qkv     = (float*)d_ws;                         // 2*576*HW
    float* dwout   = qkv     + (size_t)BATCH * 3 * DIM * HW;
    float* outv    = dwout   + (size_t)BATCH * 3 * DIM * HW;   // 2*192*HW
    float* norms   = outv    + (size_t)BATCH * DIM * HW;       // 2*384
    float* attn    = norms   + BATCH * 384;                    // 2*4*48*48
    float* partial = attn    + BATCH * NHEADS * 48 * 48;       // ATTN_KCH*2*4*2304

    // 1) qkv = w_qkv @ x  (per-batch [576x192]x[192x36864])
    gemm_wmma_kernel<<<dim3(HW / 128, (3 * DIM) / 64, BATCH), 256, 0, stream>>>(
        w_qkv, x, qkv, 3 * DIM, DIM, HW);

    // 2) depthwise 3x3
    dwconv3x3_kernel<<<dim3(HW / 256, 3 * DIM, BATCH), 256, 0, stream>>>(
        qkv, w_dw, dwout, 3 * DIM);

    // 3) inv L2 norms for q,k channels
    l2norm_kernel<<<dim3(2 * DIM, BATCH), 256, 0, stream>>>(dwout, norms);

    // 4) attn partials + 5) deterministic reduce
    attn_qk_kernel<<<dim3(ATTN_KCH, NHEADS, BATCH), 288, 0, stream>>>(
        dwout, norms, partial);
    const int attn_n = BATCH * NHEADS * 48 * 48;                // 18432
    attn_reduce_kernel<<<attn_n / 4 / 256 + 1, 256, 0, stream>>>(
        partial, attn, attn_n);

    // 6) temperature + softmax
    softmax_kernel<<<dim3(48, NHEADS, BATCH), 64, 0, stream>>>(attn, log_temp);

    // 7) out = attn @ v
    attnv_kernel<<<dim3(HW / 128, NHEADS, BATCH), 256, 0, stream>>>(
        attn, dwout, outv);

    // 8) y = w_proj @ out
    gemm_wmma_kernel<<<dim3(HW / 128, DIM / 64, BATCH), 256, 0, stream>>>(
        w_proj, outv, y, DIM, DIM, HW);
}
